// UniversalGeometricNetwork_28398323761424
// MI455X (gfx1250) — matrix operand order repaired and checked by reading, not compiled, not run on hardware
//
#include <hip/hip_runtime.h>
#include <hip/hip_bf16.h>

// ---------------------------------------------------------------------------
// MI455X (gfx1250): all eight 8192x1024x1024 GEMMs run on
// v_wmma_f32_16x16x32_bf16 (wave32, f32 accum), bf16 operands.
// Weights repacked once per launch; each (K=32, N=128) slab is 8KB contiguous.
// K-loop software-pipelined, double-buffered LDS with 80B row pitch
// (conflict-free ds_load_b128 fragments). B slab staged via
// GLOBAL_LOAD_ASYNC_TO_LDS_B128 when the builtin is available.
// Epilogues fused; all transcendental math inline (no libcalls/spills).
// ---------------------------------------------------------------------------

typedef __attribute__((ext_vector_type(16))) __bf16 bf16x16;
typedef __attribute__((ext_vector_type(8)))  float  f32x8;

#define FDIM  1024      // feature / latent dim
#define BATCH 8192
#define NB    16        // bundles
#define HH    64        // norm-mlp hidden

// LDS geometry: 80B pitch per 32-elem bf16 row (64B data + 16B pad)
#define APITCH 80
#define AREG   (128 * APITCH)          // 10240 B per A tile
#define BUFSZ  (2 * AREG)              // A + B regions per buffer

struct __align__(16) U128 { unsigned int a, b, c, d; };
struct __align__(8)  U64t { unsigned int x, y; };

#if defined(__HIP_DEVICE_COMPILE__) && \
    __has_builtin(__builtin_amdgcn_global_load_async_to_lds_b128) && \
    __has_builtin(__builtin_amdgcn_s_wait_asynccnt)
#define HAVE_ASYNC_LDS 1
typedef int v4i __attribute__((__vector_size__(16)));
typedef __attribute__((address_space(1))) v4i as1_v4i;
typedef __attribute__((address_space(3))) v4i as3_v4i;
#else
#define HAVE_ASYNC_LDS 0
#endif

// inline erf (Abramowitz-Stegun 7.1.26, |err| < 1.5e-7) -> exact-GELU flavor
__device__ __forceinline__ float gelu_exact(float x) {
    const float ax = fabsf(x) * 0.70710678118654752f;
    const float t  = 1.0f / (1.0f + 0.3275911f * ax);
    float p = 1.061405429f;
    p = p * t - 1.453152027f;
    p = p * t + 1.421413741f;
    p = p * t - 0.284496736f;
    p = p * t + 0.254829592f;
    const float erf_ax = 1.0f - p * t * __expf(-ax * ax);
    return 0.5f * x * (1.0f + copysignf(erf_ax, x));
}

// ---------------------------------------------------------------------------
// Weight packing: W[k][n] (f32, 1024x1024) -> bf16 packed:
//   dst[((k/32)*1024 + n)*32 + (k%32)] = bf16(W[k][n])
// grid = (32 k-stages, 4 n-blocks), block = 256
// ---------------------------------------------------------------------------
__global__ __launch_bounds__(256)
void pack_dense_kernel(const float* __restrict__ src, __bf16* __restrict__ dst) {
    __shared__ float tile[32 * 256];
    const int tid = threadIdx.x;
    const int s   = blockIdx.x;        // k-stage
    const int n0  = blockIdx.y * 256;  // n offset
#pragma unroll 8
    for (int c = 0; c < 32; ++c)
        tile[c * 256 + tid] = src[(long)(s * 32 + c) * FDIM + n0 + tid];
    __syncthreads();
    __bf16* base = dst + ((long)s * FDIM + n0) * 32;   // 8192 contiguous elems
#pragma unroll
    for (int it = 0; it < 4; ++it) {
        const int c   = it * 256 + tid;   // 16B chunk index, 0..1023
        const int nl  = c >> 2;           // local n
        const int kr0 = (c & 3) * 8;      // k within stage
        union { __bf16 h[8]; U128 u; } w;
#pragma unroll
        for (int q = 0; q < 8; ++q) w.h[q] = (__bf16)tile[(kr0 + q) * 256 + nl];
        ((U128*)base)[c] = w.u;
    }
}

// ---------------------------------------------------------------------------
// Mix packing: mix_w[l][i][j][kk][d] -> per-layer matrix M[k=j*64+d][n=i*64+kk]
// grid = (4096, 4 layers), block = 256; coalesced reads AND writes.
// ---------------------------------------------------------------------------
__global__ __launch_bounds__(256)
void pack_mix_kernel(const float* __restrict__ mix, __bf16* __restrict__ dstbase) {
    const int l   = blockIdx.y;
    const int idx = blockIdx.x * 256 + threadIdx.x;       // ((i*16+j)*64+kk)*64+d
    const int d   = idx & 63;
    const int kk  = (idx >> 6) & 63;
    const int j   = (idx >> 12) & 15;
    const int i   = (idx >> 16) & 15;
    const float v = mix[((long)l << 20) + idx];
    const int k = j * 64 + d;
    const int n = i * 64 + kk;
    __bf16* dst = dstbase + ((long)l << 20);
    dst[((long)(k >> 5) * FDIM + n) * 32 + (k & 31)] = (__bf16)v;
}

// ---------------------------------------------------------------------------
// WMMA GEMM, software-pipelined, double-buffered LDS (80B pitch rows).
// block tile 128x128, 8 waves (2x4), wave tile 64x32 (4x2 wmma tiles), BK=32.
// MODE 0: +bias, GELU.  MODE 1: +bias.
// MODE 2: out = zold*(1+scales[row, n/64]) + sigmoid(gateb[n/64]) * acc
// ---------------------------------------------------------------------------
template<int MODE>
__global__ __launch_bounds__(256)
void gemm_kernel(const float* __restrict__ A,
                 const __bf16* __restrict__ Bp,
                 const float* __restrict__ bias,
                 const float* __restrict__ zold,
                 const float* __restrict__ scales,
                 const float* __restrict__ gateb,
                 float* __restrict__ out)
{
    __shared__ __align__(16) unsigned char smem[2][BUFSZ];

    const int tid   = threadIdx.x;
    const int lane  = tid & 31;
    const int wave  = tid >> 5;
    const int waveM = wave >> 2;              // 0..1
    const int waveN = wave & 3;               // 0..3
    const int rowBase = blockIdx.y * 128;
    const int colBase = blockIdx.x * 128;
    const int l15 = lane & 15;
    const int lhi = lane >> 4;

    // ---- hoisted staging addresses ----
    const int c4 = (tid & 7) * 4;             // A: k offset 0..28
    const int r0 = tid >> 3;                  // A: row 0..31 (x4 passes)
    const float* gA0 = A + (long)(rowBase + r0) * FDIM + c4;
    const unsigned char* gB0 = (const unsigned char*)(Bp + (long)colBase * 32);
    const int ldsAoff = r0 * APITCH + c4 * 2;                  // bytes
    // fragment LDS byte offsets (fixed per lane)
    const int aOffB = (waveM * 64 + l15) * APITCH + lhi * 32;          // + m*16*APITCH
    const int bOffB = AREG + (waveN * 32 + l15) * APITCH + lhi * 32;   // + n*16*APITCH

    f32x8 acc[4][2] = {};

    float4 ra[4];

    auto fetchA = [&](int kt) {
#pragma unroll
        for (int p = 0; p < 4; ++p)
            ra[p] = *(const float4*)(gA0 + (long)p * 32 * FDIM + kt * 32);
    };
    auto stageA = [&](int buf) {
        unsigned char* lA = &smem[buf][0];
#pragma unroll
        for (int p = 0; p < 4; ++p) {
            union { __bf16 h[4]; U64t u; } pk;
            pk.h[0] = (__bf16)ra[p].x; pk.h[1] = (__bf16)ra[p].y;
            pk.h[2] = (__bf16)ra[p].z; pk.h[3] = (__bf16)ra[p].w;
            *(U64t*)(lA + p * 32 * APITCH + ldsAoff) = pk.u;
        }
    };

#if HAVE_ASYNC_LDS
    auto copyB = [&](int kt, int buf) {
        const unsigned char* g = gB0 + (long)kt * 65536;   // 1024*32*2 B per stage
        unsigned char* lbase = &smem[buf][AREG];
#pragma unroll
        for (int h = 0; h < 2; ++h) {
            const int c = tid + h * 256;                   // 16B chunk index
            unsigned char* l = lbase + (c >> 2) * APITCH + (c & 3) * 16;
            __builtin_amdgcn_global_load_async_to_lds_b128(
                (as1_v4i*)(unsigned long long)(g + c * 16),
                (as3_v4i*)(unsigned)(unsigned long long)l,
                0, 0);
        }
    };
#else
    U128 rb[2];
    auto fetchB = [&](int kt) {
        const unsigned char* g = gB0 + (long)kt * 65536;
#pragma unroll
        for (int h = 0; h < 2; ++h)
            rb[h] = *(const U128*)(g + (tid + h * 256) * 16);
    };
    auto stageB = [&](int buf) {
        unsigned char* lbase = &smem[buf][AREG];
#pragma unroll
        for (int h = 0; h < 2; ++h) {
            const int c = tid + h * 256;
            *(U128*)(lbase + (c >> 2) * APITCH + (c & 3) * 16) = rb[h];
        }
    };
#endif

    // ---- prologue: fill buffer 0 ----
#if HAVE_ASYNC_LDS
    copyB(0, 0);
    fetchA(0);
    stageA(0);
    __builtin_amdgcn_s_wait_asynccnt(0);
#else
    fetchA(0);
    fetchB(0);
    stageA(0);
    stageB(0);
#endif
    __syncthreads();

    for (int kt = 0; kt < FDIM / 32; ++kt) {
        const int cur = kt & 1;
        if (kt < FDIM / 32 - 1) {
#if HAVE_ASYNC_LDS
            copyB(kt + 1, cur ^ 1);    // async copy overlaps the WMMAs below
            fetchA(kt + 1);
#else
            fetchA(kt + 1);
            fetchB(kt + 1);
#endif
        }

        const unsigned char* sb = smem[cur];
        union FragU { bf16x16 v; U128 q[2]; };
        FragU fb[2];
#pragma unroll
        for (int n = 0; n < 2; ++n) {
            const unsigned char* p = sb + bOffB + n * 16 * APITCH;
            fb[n].q[0] = *(const U128*)p;
            fb[n].q[1] = *(const U128*)(p + 16);
        }
#pragma unroll
        for (int m = 0; m < 4; ++m) {
            FragU fa;
            const unsigned char* p = sb + aOffB + m * 16 * APITCH;
            fa.q[0] = *(const U128*)p;
            fa.q[1] = *(const U128*)(p + 16);
#pragma unroll
            for (int n = 0; n < 2; ++n)
                acc[m][n] = __builtin_amdgcn_wmma_f32_16x16x32_bf16(
                    false, fa.v, false, fb[n].v, (short)0, acc[m][n], false, false);
        }

        if (kt < FDIM / 32 - 1) {
            stageA(cur ^ 1);
#if !HAVE_ASYNC_LDS
            stageB(cur ^ 1);
#endif
        }
#if HAVE_ASYNC_LDS
        __builtin_amdgcn_s_wait_asynccnt(0);
#endif
        __syncthreads();
    }

    // ---- fused epilogue; C layout: VGPR v -> M = v + 8*(lane/16), N = lane%16
#pragma unroll
    for (int n = 0; n < 2; ++n) {
        const int col = colBase + waveN * 32 + n * 16 + l15;
        float bn = 0.0f, gn = 0.0f;
        if (MODE == 0 || MODE == 1) bn = bias[col];
        if (MODE == 2) gn = 1.0f / (1.0f + __expf(-gateb[col >> 6]));
#pragma unroll
        for (int m = 0; m < 4; ++m) {
            const int rbase = rowBase + waveM * 64 + m * 16 + lhi * 8;
#pragma unroll
            for (int v = 0; v < 8; ++v) {
                const int row = rbase + v;
                float x = acc[m][n][v];
                if (MODE == 0)      { x += bn; x = gelu_exact(x); }
                else if (MODE == 1) { x += bn; }
                else {
                    const float zo = zold[(long)row * FDIM + col];
                    const float sc = scales[(long)row * NB + (col >> 6)];
                    x = zo * (1.0f + sc) + gn * x;
                }
                out[(long)row * FDIM + col] = x;
            }
        }
    }
}

// ---------------------------------------------------------------------------
// Per-row bundle MLP: norms -> gelu(nw1+b1) -> gelu(hw2+b2) -> softplus(hw3)
// One wave per batch row; shuffle reductions for the 16 bundle norms.
// grid = 1024, block = 256 (8 rows/block)
// ---------------------------------------------------------------------------
__global__ __launch_bounds__(256)
void bundle_mlp_kernel(const float* __restrict__ z,
                       const float* __restrict__ w1, const float* __restrict__ b1,
                       const float* __restrict__ w2, const float* __restrict__ b2,
                       const float* __restrict__ w3,
                       float* __restrict__ scales)
{
    __shared__ float sh1[8][64];
    __shared__ float sh2[8][64];
    const int lane = threadIdx.x & 31;
    const int wave = threadIdx.x >> 5;
    const int row  = blockIdx.x * 8 + wave;

    float xv[32];
    const float* zr = z + (long)row * FDIM;
#pragma unroll
    for (int j = 0; j < 32; ++j) xv[j] = zr[j * 32 + lane];

    float norms[NB];
#pragma unroll
    for (int n = 0; n < NB; ++n) {
        float p = xv[2 * n] * xv[2 * n] + xv[2 * n + 1] * xv[2 * n + 1];
#pragma unroll
        for (int off = 16; off > 0; off >>= 1) p += __shfl_xor(p, off);
        norms[n] = sqrtf(p) + 1e-8f;
    }

    float h1a = b1[lane], h1b = b1[lane + 32];
#pragma unroll
    for (int n = 0; n < NB; ++n) {
        h1a += norms[n] * w1[n * HH + lane];
        h1b += norms[n] * w1[n * HH + lane + 32];
    }
    sh1[wave][lane]      = gelu_exact(h1a);
    sh1[wave][lane + 32] = gelu_exact(h1b);
    __syncthreads();

    float h2a = b2[lane], h2b = b2[lane + 32];
#pragma unroll 8
    for (int i = 0; i < HH; ++i) {
        const float hv = sh1[wave][i];
        h2a += hv * w2[i * HH + lane];
        h2b += hv * w2[i * HH + lane + 32];
    }
    sh2[wave][lane]      = gelu_exact(h2a);
    sh2[wave][lane + 32] = gelu_exact(h2b);
    __syncthreads();

    if (lane < NB) {
        float s = 0.0f;
#pragma unroll 8
        for (int i = 0; i < HH; ++i) s += sh2[wave][i] * w3[i * NB + lane];
        s = (s > 20.0f) ? s : __logf(1.0f + __expf(s));   // softplus
        scales[(long)row * NB + lane] = s;
    }
}

// ---------------------------------------------------------------------------
extern "C" void kernel_launch(void* const* d_in, const int* in_sizes, int n_in,
                              void* d_out, int out_size, void* d_ws, size_t ws_size,
                              hipStream_t stream) {
    (void)in_sizes; (void)n_in; (void)out_size; (void)ws_size;
    const float* x      = (const float*)d_in[0];
    const float* enc_w1 = (const float*)d_in[1];
    const float* enc_b1 = (const float*)d_in[2];
    const float* enc_w2 = (const float*)d_in[3];
    const float* enc_b2 = (const float*)d_in[4];
    const float* mlp_w1 = (const float*)d_in[5];
    const float* mlp_b1 = (const float*)d_in[6];
    const float* mlp_w2 = (const float*)d_in[7];
    const float* mlp_b2 = (const float*)d_in[8];
    const float* mlp_w3 = (const float*)d_in[9];
    const float* mix_w  = (const float*)d_in[10];
    const float* gate_b = (const float*)d_in[11];
    const float* dec_w1 = (const float*)d_in[12];
    const float* dec_b1 = (const float*)d_in[13];
    const float* dec_w2 = (const float*)d_in[14];
    const float* dec_b2 = (const float*)d_in[15];

    char*   ws   = (char*)d_ws;
    __bf16* pw   = (__bf16*)ws;                                 // 8 x 2MB packed
    float*  bufA = (float*)(ws + (size_t)16 * 1024 * 1024);     // 32MB
    float*  bufB = (float*)(ws + (size_t)48 * 1024 * 1024);     // 32MB
    float*  scl  = (float*)(ws + (size_t)80 * 1024 * 1024);     // 512KB

    const size_t MAT = (size_t)1 << 20;   // elems per packed matrix

    dim3 pgrid(32, 4);
    pack_dense_kernel<<<pgrid, 256, 0, stream>>>(enc_w1, pw + 0 * MAT);
    pack_dense_kernel<<<pgrid, 256, 0, stream>>>(enc_w2, pw + 1 * MAT);
    pack_dense_kernel<<<pgrid, 256, 0, stream>>>(dec_w1, pw + 6 * MAT);
    pack_dense_kernel<<<pgrid, 256, 0, stream>>>(dec_w2, pw + 7 * MAT);
    pack_mix_kernel<<<dim3(4096, 4), 256, 0, stream>>>(mix_w, pw + 2 * MAT);

    dim3 ggrid(FDIM / 128, BATCH / 128);   // (8, 64)
    gemm_kernel<0><<<ggrid, 256, 0, stream>>>(x,    pw + 0 * MAT, enc_b1, nullptr, nullptr, nullptr, bufA);
    gemm_kernel<1><<<ggrid, 256, 0, stream>>>(bufA, pw + 1 * MAT, enc_b2, nullptr, nullptr, nullptr, bufB);

    float* zin = bufB; float* zout = bufA;
    for (int l = 0; l < 4; ++l) {
        bundle_mlp_kernel<<<BATCH / 8, 256, 0, stream>>>(
            zin, mlp_w1 + l * NB * HH, mlp_b1 + l * HH,
            mlp_w2 + l * HH * HH, mlp_b2 + l * HH,
            mlp_w3 + l * HH * NB, scl);
        gemm_kernel<2><<<ggrid, 256, 0, stream>>>(
            zin, pw + (2 + l) * MAT, nullptr, zin, scl, gate_b + l * NB, zout);
        float* t = zin; zin = zout; zout = t;
    }
    // after 4 swaps: zin == bufB, zout == bufA
    gemm_kernel<0><<<ggrid, 256, 0, stream>>>(zin,  pw + 6 * MAT, dec_b1, nullptr, nullptr, nullptr, zout);
    gemm_kernel<1><<<ggrid, 256, 0, stream>>>(zout, pw + 7 * MAT, dec_b2, nullptr, nullptr, nullptr, (float*)d_out);
}
